// SAGE_66185446031414
// MI455X (gfx1250) — matrix operand order repaired and checked by reading, not compiled, not run on hardware
//
#include <hip/hip_runtime.h>
#include <hip/hip_bf16.h>

#define DIM 128
#define NGRAPH 128

typedef __attribute__((ext_vector_type(16))) __bf16 v16bf;
typedef __attribute__((ext_vector_type(8)))  float  v8f;

// granule-XOR swizzle for transposed bf16 weights in LDS:
// element (col c, k) of W^T stored at c*128 + (((k>>4)^c)&7)*16 + (k&15)
__device__ __forceinline__ int swz(int c, int k) {
    return c * DIM + ((((k >> 4) ^ c) & 7) << 4) + (k & 15);
}

__global__ void zero_kernel(float* __restrict__ p, long n) {
    long i = (long)blockIdx.x * blockDim.x + threadIdx.x;
    long s = (long)gridDim.x * blockDim.x;
    for (; i < n; i += s) p[i] = 0.0f;
}

__global__ void degree_kernel(const int* __restrict__ dst, int E, float* __restrict__ deg) {
    int i = blockIdx.x * blockDim.x + threadIdx.x;
    int s = gridDim.x * blockDim.x;
    for (; i < E; i += s) atomicAdd(&deg[dst[i]], 1.0f);
}

__global__ void deg_inv_kernel(float* __restrict__ deg, int N) {
    int i = blockIdx.x * blockDim.x + threadIdx.x;
    if (i < N) {
        float d = deg[i];
        deg[i] = (d > 0.0f) ? (1.0f / d) : 0.0f;
    }
}

// one wave32 per edge: lane l covers feature columns [4l, 4l+4)
__global__ void aggregate_kernel(const float* __restrict__ hin,
                                 const int* __restrict__ src,
                                 const int* __restrict__ dst,
                                 float* __restrict__ agg, int E) {
    int gt   = blockIdx.x * blockDim.x + threadIdx.x;
    int w    = gt >> 5;
    int lane = gt & 31;
    int nw   = (gridDim.x * blockDim.x) >> 5;
    for (int e = w; e < E; e += nw) {
        int s = src[e];
        int d = dst[e];
        const float4 v = *(const float4*)(hin + (size_t)s * DIM + lane * 4);
        float* ap = agg + (size_t)d * DIM + lane * 4;
        atomicAdd(ap + 0, v.x);
        atomicAdd(ap + 1, v.y);
        atomicAdd(ap + 2, v.z);
        atomicAdd(ap + 3, v.w);
    }
}

// Hout[16-row tile] = relu( Hin@Ws + (Agg*deg_inv)@Wn + bias ), bf16 WMMA, f32 accum.
// 8 waves/block; EACH WAVE OWNS EXACTLY ONE M-tile (no tile loop -> no LICM
// hoisting of the LDS B-fragment loads -> no scratch spills).
__launch_bounds__(256)
__global__ void sage_gemm_kernel(const float* __restrict__ Hin,
                                 const float* __restrict__ Agg,
                                 const float* __restrict__ deg_inv,
                                 const float* __restrict__ Ws,
                                 const float* __restrict__ Wn,
                                 const float* __restrict__ bias,
                                 float* __restrict__ Hout, int N) {
    __shared__ __align__(32) __bf16 ldsWs[DIM * DIM];
    __shared__ __align__(32) __bf16 ldsWn[DIM * DIM];

    // convert W (row-major [k][c], fp32) -> LDS W^T (bf16, swizzled)
    for (int i = threadIdx.x; i < DIM * DIM; i += blockDim.x) {
        int k = i >> 7, c = i & 127;
        int a = swz(c, k);
        ldsWs[a] = (__bf16)Ws[i];
        ldsWn[a] = (__bf16)Wn[i];
    }
    __syncthreads();

    const int lane = threadIdx.x & 31;
    const int wave = threadIdx.x >> 5;
    const int m    = lane & 15;   // row-in-tile / col-in-Ntile
    const int hs   = lane >> 4;   // half-wave select
    const int nTiles = (N + 15) >> 4;
    const int tile   = blockIdx.x * 8 + wave;
    if (tile >= nTiles) return;

    const int row0 = tile << 4;
    int r = row0 + m;
    if (r >= N) r = N - 1;
    const float di = deg_inv[r];

    v8f acc[8];
#pragma unroll
    for (int nt = 0; nt < 8; ++nt) {
        float bv = bias[(nt << 4) + m];
#pragma unroll
        for (int i = 0; i < 8; ++i) acc[nt][i] = bv;
    }

#pragma unroll
    for (int kc = 0; kc < 4; ++kc) {
        // A fragment (contiguous K order, same as B): half-wave 0 -> K {0..15}, hw1 -> K {16..31}
        const int ka = kc * 32 + hs * 16;
        const int g  = kc * 2 + hs;  // B: 16-element K-granule index in the row

        // ---- self term: A from Hin ----
        {
            const float* hp = Hin + (size_t)r * DIM + ka;
            v16bf a_s;
#pragma unroll
            for (int j = 0; j < 8; ++j) {
                a_s[j]     = (__bf16)hp[j];
                a_s[j + 8] = (__bf16)hp[j + 8];
            }
#pragma unroll
            for (int nt = 0; nt < 8; ++nt) {
                int c = (nt << 4) + m;
                int addr = c * DIM + (((g ^ c) & 7) << 4);
                v16bf bs = *(const v16bf*)&ldsWs[addr];
                acc[nt] = __builtin_amdgcn_wmma_f32_16x16x32_bf16(
                    false, a_s, false, bs, (short)0, acc[nt], false, false);
            }
        }
        // ---- neighbor term: A from Agg * deg_inv ----
        {
            const float* ap = Agg + (size_t)r * DIM + ka;
            v16bf a_n;
#pragma unroll
            for (int j = 0; j < 8; ++j) {
                a_n[j]     = (__bf16)(ap[j] * di);
                a_n[j + 8] = (__bf16)(ap[j + 8] * di);
            }
#pragma unroll
            for (int nt = 0; nt < 8; ++nt) {
                int c = (nt << 4) + m;
                int addr = c * DIM + (((g ^ c) & 7) << 4);
                v16bf bn = *(const v16bf*)&ldsWn[addr];
                acc[nt] = __builtin_amdgcn_wmma_f32_16x16x32_bf16(
                    false, a_n, false, bn, (short)0, acc[nt], false, false);
            }
        }
    }

    // C layout: VGPR i, halves -> row = row0 + hs*8 + i, col = nt*16 + m; fuse ReLU
#pragma unroll
    for (int i = 0; i < 8; ++i) {
        int row = row0 + hs * 8 + i;
        float* op = Hout + (size_t)row * DIM + m;
#pragma unroll
        for (int nt = 0; nt < 8; ++nt) {
            float v = acc[nt][i];
            op[nt << 4] = (v > 0.0f) ? v : 0.0f;
        }
    }
}

// one wave32 per node: accumulate per-graph feature sums + counts
__global__ void graph_accum_kernel(const float* __restrict__ h,
                                   const int* __restrict__ gid, int N,
                                   float* __restrict__ gsum, float* __restrict__ gcnt) {
    int gt   = blockIdx.x * blockDim.x + threadIdx.x;
    int w    = gt >> 5;
    int lane = gt & 31;
    int nw   = (gridDim.x * blockDim.x) >> 5;
    for (int n = w; n < N; n += nw) {
        int g = gid[n];
        const float4 v = *(const float4*)(h + (size_t)n * DIM + lane * 4);
        float* p = gsum + (size_t)g * DIM + lane * 4;
        atomicAdd(p + 0, v.x);
        atomicAdd(p + 1, v.y);
        atomicAdd(p + 2, v.z);
        atomicAdd(p + 3, v.w);
        if (lane == 0) atomicAdd(&gcnt[g], 1.0f);
    }
}

__global__ void graph_final_kernel(const float* __restrict__ gsum,
                                   const float* __restrict__ gcnt,
                                   float* __restrict__ out) {
    int i = blockIdx.x * blockDim.x + threadIdx.x;
    if (i < NGRAPH * DIM) {
        float c = gcnt[i >> 7];
        out[i] = gsum[i] / ((c > 1.0f) ? c : 1.0f);
    }
}

extern "C" void kernel_launch(void* const* d_in, const int* in_sizes, int n_in,
                              void* d_out, int out_size, void* d_ws, size_t ws_size,
                              hipStream_t stream) {
    const float* h   = (const float*)d_in[0];
    const float* W1s = (const float*)d_in[1];
    const float* W1n = (const float*)d_in[2];
    const float* b1  = (const float*)d_in[3];
    const float* W2s = (const float*)d_in[4];
    const float* W2n = (const float*)d_in[5];
    const float* b2  = (const float*)d_in[6];
    const int* src   = (const int*)d_in[7];
    const int* dst   = (const int*)d_in[8];
    const int* gid   = (const int*)d_in[9];
    float* out = (float*)d_out;

    const int N = in_sizes[0] / DIM;   // 100000
    const int E = in_sizes[7];         // 1600000
    const int nTiles = (N + 15) / 16;  // 6250
    const int gemmBlocks = (nTiles + 7) / 8;  // 782, one M-tile per wave

    // workspace layout (floats): [deg/deg_inv N][agg N*D][gsum G*D][gcnt G][h1 N*D]
    float* deg  = (float*)d_ws;
    float* agg  = deg + N;
    float* gsum = agg + (size_t)N * DIM;
    float* gcnt = gsum + (size_t)NGRAPH * DIM;
    float* h1   = gcnt + NGRAPH;

    // zero [deg .. gcnt] in one contiguous pass (h1 is fully overwritten before read)
    long zn = (long)N + (long)N * DIM + (long)NGRAPH * DIM + NGRAPH;
    zero_kernel<<<2048, 256, 0, stream>>>(deg, zn);

    degree_kernel<<<1024, 256, 0, stream>>>(dst, E, deg);
    deg_inv_kernel<<<(N + 255) / 256, 256, 0, stream>>>(deg, N);

    // layer 1
    aggregate_kernel<<<4096, 256, 0, stream>>>(h, src, dst, agg, E);
    sage_gemm_kernel<<<gemmBlocks, 256, 0, stream>>>(h, agg, deg, W1s, W1n, b1, h1, N);

    // layer 2 (in-place on h1)
    zero_kernel<<<2048, 256, 0, stream>>>(agg, (long)N * DIM);
    aggregate_kernel<<<4096, 256, 0, stream>>>(h1, src, dst, agg, E);
    sage_gemm_kernel<<<gemmBlocks, 256, 0, stream>>>(h1, agg, deg, W2s, W2n, b2, h1, N);

    // per-graph mean readout
    graph_accum_kernel<<<2048, 256, 0, stream>>>(h1, gid, N, gsum, gcnt);
    graph_final_kernel<<<(NGRAPH * DIM + 255) / 256, 256, 0, stream>>>(gsum, gcnt, out);
}
